// GraphAttentionHead_18090402250828
// MI455X (gfx1250) — compile-verified, matrix operand order checked by reading, not run on hardware
//
#include <hip/hip_runtime.h>

#define DWIDTH 512
#define SLOPE  0.2f

typedef __attribute__((ext_vector_type(16))) __bf16 v16bf;
typedef __attribute__((ext_vector_type(8)))  __bf16 v8bf;
typedef __attribute__((ext_vector_type(8)))  float  v8f;

// ---------------------------------------------------------------- utilities
__global__ void zero_f32(float* __restrict__ p, int n) {
  int i = blockIdx.x * blockDim.x + threadIdx.x;
  if (i < n) p[i] = 0.0f;
}

__global__ void cvt_bf16(const float* __restrict__ src, __bf16* __restrict__ dst, int n) {
  int i = blockIdx.x * blockDim.x + threadIdx.x;
  if (i < n) dst[i] = (__bf16)src[i];
}

// W is [K=512][N=512] row-major; produce Wt [N][K] in bf16 so B fragments are
// contiguous 16-byte loads per lane.
__global__ void cvt_bf16_T512(const float* __restrict__ W, __bf16* __restrict__ Wt) {
  int i = blockIdx.x * blockDim.x + threadIdx.x;   // i < 512*512
  int k = i >> 9;
  int n = i & 511;
  Wt[(size_t)n * DWIDTH + k] = (__bf16)W[i];
}

// ---------------------------------------------------------------- GEMM: h = A @ W + b
// One wave computes one 16x16 tile of H. A: [N][512] bf16 row-major.
// Bt: [512][512] bf16 = W transposed (column n of W is row n of Bt).
__global__ __launch_bounds__(256)
void gat_gemm_wmma(const __bf16* __restrict__ A, const __bf16* __restrict__ Bt,
                   const float* __restrict__ bw, float* __restrict__ H,
                   int mTiles) {
  const int wave = (blockIdx.x * (blockDim.x >> 5)) + (threadIdx.x >> 5);
  const int lane = threadIdx.x & 31;
  const int tile_m = wave >> 5;        // 32 column tiles (512/16)
  const int tile_n = wave & 31;
  if (tile_m >= mTiles) return;        // wave-uniform; EXEC stays all-1s

  const int hf = lane >> 4;            // 0: K 0..7 / 16..23, 1: K 8..15 / 24..31
  const int lm = lane & 15;            // row (A) / col (B) within tile

  const __bf16* arow = A  + (size_t)(tile_m * 16 + lm) * DWIDTH;
  const __bf16* brow = Bt + (size_t)(tile_n * 16 + lm) * DWIDTH;

  v8f acc = {};
#pragma unroll 4
  for (int k = 0; k < DWIDTH; k += 32) {
    const int k0 = k + hf * 8;
    v8bf a0 = *(const v8bf*)(arow + k0);
    v8bf a1 = *(const v8bf*)(arow + k0 + 16);
    v8bf b0 = *(const v8bf*)(brow + k0);
    v8bf b1 = *(const v8bf*)(brow + k0 + 16);
    v16bf av = __builtin_shufflevector(a0, a1, 0,1,2,3,4,5,6,7,8,9,10,11,12,13,14,15);
    v16bf bv = __builtin_shufflevector(b0, b1, 0,1,2,3,4,5,6,7,8,9,10,11,12,13,14,15);
    acc = __builtin_amdgcn_wmma_f32_16x16x32_bf16(
        /*neg_a=*/false, av, /*neg_b=*/false, bv,
        /*c_mod=*/(short)0, acc, /*reuse_a=*/false, /*reuse_b=*/false);
  }

  // C/D layout: VGPR r -> M = r + hf*8, N = lm
  const float bias = bw[tile_n * 16 + lm];
  float* hout = H + (size_t)(tile_m * 16 + hf * 8) * DWIDTH + tile_n * 16 + lm;
#pragma unroll
  for (int r = 0; r < 8; ++r)
    hout[(size_t)r * DWIDTH] = acc[r] + bias;
}

// ---------------------------------------------------------------- per-node scores
// s_dst[i] = h[i] . a[0:512], s_src[i] = h[i] . a[512:1024]; one wave per node.
__global__ __launch_bounds__(256)
void gat_scores(const float* __restrict__ H, const float* __restrict__ a,
                float* __restrict__ sd, float* __restrict__ ss, int N) {
  const int node = (blockIdx.x * (blockDim.x >> 5)) + (threadIdx.x >> 5);
  const int lane = threadIdx.x & 31;
  if (node >= N) return;
  const float4* hp = (const float4*)(H + (size_t)node * DWIDTH);
  const float4* a0 = (const float4*)a;
  const float4* a1 = (const float4*)(a + DWIDTH);
  float p0 = 0.0f, p1 = 0.0f;
#pragma unroll
  for (int j = 0; j < 4; ++j) {
    float4 hv = hp[lane * 4 + j];
    float4 u  = a0[lane * 4 + j];
    float4 v  = a1[lane * 4 + j];
    p0 += hv.x * u.x + hv.y * u.y + hv.z * u.z + hv.w * u.w;
    p1 += hv.x * v.x + hv.y * v.y + hv.z * v.z + hv.w * v.w;
  }
#pragma unroll
  for (int o = 16; o > 0; o >>= 1) {
    p0 += __shfl_xor(p0, o, 32);
    p1 += __shfl_xor(p1, o, 32);
  }
  if (lane == 0) { sd[node] = p0; ss[node] = p1; }
}

// ---------------------------------------------------------------- per-edge softmax numerator
__global__ void gat_edges(const int* __restrict__ ei, int E,
                          const float* __restrict__ sd, const float* __restrict__ ss,
                          const float* __restrict__ ba,
                          float* __restrict__ wE, float* __restrict__ denom) {
  int e = blockIdx.x * blockDim.x + threadIdx.x;
  if (e >= E) return;
  int s = ei[e];        // src (neighbor j)
  int d = ei[E + e];    // dst (center i)
  float t  = sd[d] + ss[s] + ba[0];
  float lr = t > 0.0f ? t : SLOPE * t;
  float w  = __expf(lr);
  wE[e] = w;
  atomicAdd(&denom[d], w);
}

// ---------------------------------------------------------------- weighted scatter
// One wave per edge; each lane accumulates 16 features of out[dst] += alpha*h[src].
__global__ __launch_bounds__(256)
void gat_scatter(const int* __restrict__ ei, int E,
                 const float* __restrict__ wE, const float* __restrict__ denom,
                 const float* __restrict__ H, float* __restrict__ out) {
  const int e = (blockIdx.x * (blockDim.x >> 5)) + (threadIdx.x >> 5);
  const int lane = threadIdx.x & 31;
  if (e >= E) return;
  const int s = ei[e];
  const int d = ei[E + e];
  const float alpha = wE[e] / denom[d];
  const float4* hs = (const float4*)(H + (size_t)s * DWIDTH);
  float* od = out + (size_t)d * DWIDTH;
#pragma unroll
  for (int j = 0; j < 4; ++j) {
    float4 v = hs[lane * 4 + j];
    int b = (lane * 4 + j) * 4;
    unsafeAtomicAdd(od + b + 0, alpha * v.x);   // global_atomic_add_f32
    unsafeAtomicAdd(od + b + 1, alpha * v.y);
    unsafeAtomicAdd(od + b + 2, alpha * v.z);
    unsafeAtomicAdd(od + b + 3, alpha * v.w);
  }
}

// ---------------------------------------------------------------- launch
extern "C" void kernel_launch(void* const* d_in, const int* in_sizes, int n_in,
                              void* d_out, int out_size, void* d_ws, size_t ws_size,
                              hipStream_t stream) {
  const float* nodes = (const float*)d_in[0];
  const int*   ei    = (const int*)d_in[1];
  const float* W     = (const float*)d_in[2];
  const float* bw    = (const float*)d_in[3];
  const float* a     = (const float*)d_in[4];
  const float* ba    = (const float*)d_in[5];
  float* out = (float*)d_out;

  const int N = in_sizes[0] / DWIDTH;   // 20000
  const int E = in_sizes[1] / 2;        // 160000
  (void)n_in; (void)out_size; (void)ws_size;

  // workspace carve-out (256B aligned)
  char* ws = (char*)d_ws;
  size_t off = 0;
  auto take = [&](size_t bytes) -> char* {
    char* p = ws + off;
    off = (off + bytes + 255) & ~(size_t)255;
    return p;
  };
  float*  H   = (float*)take((size_t)N * DWIDTH * sizeof(float));    // 40.96 MB
  __bf16* Ab  = (__bf16*)take((size_t)N * DWIDTH * sizeof(__bf16));  // 20.48 MB
  __bf16* Wt  = (__bf16*)take((size_t)DWIDTH * DWIDTH * sizeof(__bf16));
  float*  sd  = (float*)take((size_t)N * sizeof(float));
  float*  ssc = (float*)take((size_t)N * sizeof(float));
  float*  den = (float*)take((size_t)N * sizeof(float));
  float*  wE  = (float*)take((size_t)E * sizeof(float));

  // 1) zero accumulators (d_out / d_ws are poisoned by the harness)
  zero_f32<<<(N * DWIDTH + 255) / 256, 256, 0, stream>>>(out, N * DWIDTH);
  zero_f32<<<(N + 255) / 256, 256, 0, stream>>>(den, N);

  // 2) fp32 -> bf16 conversions (W transposed for contiguous B fragments)
  cvt_bf16<<<(N * DWIDTH + 255) / 256, 256, 0, stream>>>(nodes, Ab, N * DWIDTH);
  cvt_bf16_T512<<<(DWIDTH * DWIDTH + 255) / 256, 256, 0, stream>>>(W, Wt);

  // 3) h = nodes @ W + b  (WMMA bf16, fp32 accumulate)
  const int mTiles = (N + 15) / 16;            // 1250
  const int nTiles = DWIDTH / 16;              // 32
  const int waves  = mTiles * nTiles;          // 40000 -> 5000 blocks of 8 waves
  gat_gemm_wmma<<<(waves + 7) / 8, 256, 0, stream>>>(Ab, Wt, bw, H, mTiles);

  // 4) per-node attention partial scores
  gat_scores<<<(N + 7) / 8, 256, 0, stream>>>(H, a, sd, ssc, N);

  // 5) per-edge exp(leaky_relu) + denominator
  gat_edges<<<(E + 255) / 256, 256, 0, stream>>>(ei, E, sd, ssc, ba, wE, den);

  // 6) out[dst] += (w/denom[dst]) * h[src]
  gat_scatter<<<(E + 7) / 8, 256, 0, stream>>>(ei, E, wE, den, H, out);
}
